// AlignmentAttention_82755429860169
// MI455X (gfx1250) — matrix operand order repaired and checked by reading, not compiled, not run on hardware
//
#include <hip/hip_runtime.h>
#include <hip/hip_bf16.h>
#include <math.h>

// ---------------- CDNA5 (gfx1250) types for WMMA ----------------
typedef __attribute__((ext_vector_type(16))) __bf16 v16bf;
typedef __attribute__((ext_vector_type(8)))  float  v8f;

#define E_DIM 512
#define S_LEN 256
#define NB    512
#define DIM   256
#define HID   512
#define RHO_C 0.5f
#define NEG_S 0.01f

struct Frag32B { uint4 lo; uint4 hi; };   // 32 bytes == v16bf

__device__ __forceinline__ unsigned short f2bf(float f) {
  // round-to-nearest-even float -> bf16
  unsigned u = __float_as_uint(f);
  unsigned r = u + 0x7FFFu + ((u >> 16) & 1u);
  return (unsigned short)(r >> 16);
}
__device__ __forceinline__ float bf2f(unsigned short h) {
  return __uint_as_float(((unsigned)h) << 16);
}

// Async global->LDS copy of 16B per lane (GLOBAL_LOAD_ASYNC_TO_LDS_B128,
// tracked by ASYNCcnt).  ldsoff = byte offset in the workgroup LDS
// allocation (low 32 bits of the flat LDS address per the aperture rule).
__device__ __forceinline__ void async_copy_b128(unsigned ldsoff, const void* gsrc) {
  asm volatile("global_load_async_to_lds_b128 %0, %1, off"
               :: "v"(ldsoff), "v"((unsigned long long)(uintptr_t)gsrc)
               : "memory");
}
__device__ __forceinline__ void wait_async0() {
  asm volatile("s_wait_asynccnt 0x0" ::: "memory");
}

// ---------------------------------------------------------------------------
// Kernel 0: convert WT/WH to bf16 (row-major [E,E]) and zero the I accumulator.
// ---------------------------------------------------------------------------
__global__ __launch_bounds__(256)
void cvt_weights(const float* __restrict__ WT, const float* __restrict__ WH,
                 unsigned short* __restrict__ wtb, unsigned short* __restrict__ whb,
                 float* __restrict__ I, int iElems) {
  int idx = blockIdx.x * blockDim.x + threadIdx.x;
  int stride = gridDim.x * blockDim.x;
  const int n = E_DIM * E_DIM;
  for (int i = idx; i < n; i += stride) {
    wtb[i] = f2bf(WT[i]);
    whb[i] = f2bf(WH[i]);
  }
  for (int i = idx; i < iElems; i += stride) I[i] = 0.0f;
}

// ---------------------------------------------------------------------------
// Kernel 1: fused critic phase-1.
//   For batch b (b==NB -> Q), n-tile of 64 output columns:
//     T = X_b @ WT^T ; Hh = X_b @ WH^T   (bf16 WMMA, f32 accum)
//     I[b,col] += sum_s relu(Hh)*sigmoid(T) + x*(1-sigmoid(T))
// Block = 256 threads = 8 waves.  Wave (mwave 0..3, ngrp 0..1).
// Weight tiles (64 cols x 32 K x {WT,WH}) are double-buffered in LDS via
// GLOBAL_LOAD_ASYNC_TO_LDS_B128, prefetched one k-step ahead.
// ---------------------------------------------------------------------------
__global__ __launch_bounds__(256)
void critic_phase1(const float* __restrict__ K, const float* __restrict__ Q,
                   const unsigned short* __restrict__ wtb,
                   const unsigned short* __restrict__ whb,
                   const float* __restrict__ biasT, const float* __restrict__ biasH,
                   float* __restrict__ I) {
  __shared__ unsigned int sU[64 * 256];  // X tile: 64 rows x 512 cols bf16, 64 KB
  // Weight slab: 2 buffers x 2 matrices x 64 rows x 80B (64B data + 16B pad)
  __shared__ unsigned int sW[2 * 2 * 64 * 20];   // 20 KB

  const int b    = blockIdx.x;
  const int n0   = blockIdx.y * 64;
  const int tid  = threadIdx.x;
  const int lane = tid & 31;
  const int wave = tid >> 5;
  const int mwave = wave & 3;      // M subtile within 64-row pass
  const int ngrp  = wave >> 2;     // 0..1 -> n-subtile pair
  const int lhalf = lane >> 4;     // 0 | 1 (half-wave, per ISA fragment layouts)
  const int l15   = lane & 15;

  const float* xp = (b < NB) ? (K + (size_t)b * (S_LEN * E_DIM)) : Q;

  // per-thread async-copy assignment: 2 x b128 per thread per tile
  // q = tid + 256*i : mat = q>>8, row = (q>>2)&63, seg = q&3
  const unsigned swbase = (unsigned)(uintptr_t)&sW[0];

  for (int pass = 0; pass < 4; ++pass) {
    const int m0 = pass * 64;
    __syncthreads();   // previous pass finished all sU / sW reads

    // ---- prefetch weight tile for kk=0 into buffer 0 (overlaps X staging) ----
    #pragma unroll
    for (int i = 0; i < 2; ++i) {
      const int q   = tid + 256 * i;
      const int mat = q >> 8;
      const int row = (q >> 2) & 63;
      const int seg = q & 3;
      const unsigned short* gsrc =
          (mat ? whb : wtb) + (size_t)(n0 + row) * E_DIM + /*k0=*/0 + seg * 8;
      const unsigned off = swbase + (unsigned)((mat * 1280 + row * 20 + seg * 4) * 4);
      async_copy_b128(off, gsrc);
    }

    // ---- stage X[m0..m0+63, 0..511] -> LDS bf16 (8192 float4 / 256 threads) ----
    const float4* src = (const float4*)(xp + (size_t)m0 * E_DIM);
    #pragma unroll 4
    for (int it = 0; it < 32; ++it) {
      int q = tid + 256 * it;
      float4 v = src[q];
      sU[q * 2 + 0] = ((unsigned)f2bf(v.y) << 16) | (unsigned)f2bf(v.x);
      sU[q * 2 + 1] = ((unsigned)f2bf(v.w) << 16) | (unsigned)f2bf(v.z);
    }
    if (pass < 3) __builtin_prefetch(xp + (size_t)(m0 + 64) * E_DIM, 0, 0);
    __syncthreads();

    v8f accT0 = {}, accT1 = {}, accH0 = {}, accH1 = {};
    const int mrel  = mwave * 16;
    const int c0rel = (ngrp * 2 + 0) * 16 + l15;   // column within 64-col tile
    const int c1rel = (ngrp * 2 + 1) * 16 + l15;
    const int c0 = n0 + c0rel;
    const int c1 = n0 + c1rel;

    for (int kk = 0; kk < 16; ++kk) {
      const int p = kk & 1;
      wait_async0();        // this wave's copies for tile kk have landed
      __syncthreads();      // all waves' copies landed; prev parity fully read

      // prefetch tile kk+1 into the other parity (overlaps WMMA below)
      if (kk < 15) {
        const int k1 = (kk + 1) * 32;
        const int p1 = (kk + 1) & 1;
        #pragma unroll
        for (int i = 0; i < 2; ++i) {
          const int q   = tid + 256 * i;
          const int mat = q >> 8;
          const int row = (q >> 2) & 63;
          const int seg = q & 3;
          const unsigned short* gsrc =
              (mat ? whb : wtb) + (size_t)(n0 + row) * E_DIM + k1 + seg * 8;
          const unsigned off =
              swbase + (unsigned)(((p1 * 2 + mat) * 1280 + row * 20 + seg * 4) * 4);
          async_copy_b128(off, gsrc);
        }
      }

      const int k0 = kk * 32;

      // A fragment (16x32 bf16): lane<16 row=l15 K0..15; lane>=16 K16..31
      const int arow  = mrel + l15;
      const int abase = arow * 256 + ((k0 + lhalf * 16) >> 1);  // uint index into sU
      Frag32B fa;
      fa.lo = *(const uint4*)&sU[abase];
      fa.hi = *(const uint4*)&sU[abase + 4];                    // +8 bf16 elements
      v16bf a = __builtin_bit_cast(v16bf, fa);

      // B fragments (32x16 bf16) from the LDS weight slab:
      // lane = column; lane<16 reads K0..15 (bytes 0..31), lane>=16 K16..31 (bytes 32..63)
      const int baseT = (p * 2 + 0) * 1280;
      const int baseH = (p * 2 + 1) * 1280;
      const int ko = lhalf * 8;                 // 8 uints = 32B
      Frag32B ft0, ft1, fh0, fh1;
      { const uint4* s = (const uint4*)&sW[baseT + c0rel * 20 + ko]; ft0.lo = s[0]; ft0.hi = s[1]; }
      { const uint4* s = (const uint4*)&sW[baseT + c1rel * 20 + ko]; ft1.lo = s[0]; ft1.hi = s[1]; }
      { const uint4* s = (const uint4*)&sW[baseH + c0rel * 20 + ko]; fh0.lo = s[0]; fh0.hi = s[1]; }
      { const uint4* s = (const uint4*)&sW[baseH + c1rel * 20 + ko]; fh1.lo = s[0]; fh1.hi = s[1]; }
      v16bf bt0 = __builtin_bit_cast(v16bf, ft0);
      v16bf bt1 = __builtin_bit_cast(v16bf, ft1);
      v16bf bh0 = __builtin_bit_cast(v16bf, fh0);
      v16bf bh1 = __builtin_bit_cast(v16bf, fh1);

      accT0 = __builtin_amdgcn_wmma_f32_16x16x32_bf16(false, a, false, bt0, (short)0, accT0, false, false);
      accT1 = __builtin_amdgcn_wmma_f32_16x16x32_bf16(false, a, false, bt1, (short)0, accT1, false, false);
      accH0 = __builtin_amdgcn_wmma_f32_16x16x32_bf16(false, a, false, bh0, (short)0, accH0, false, false);
      accH1 = __builtin_amdgcn_wmma_f32_16x16x32_bf16(false, a, false, bh1, (short)0, accH1, false, false);
    }

    // ---- fused gating epilogue + sequence-sum ----
    // D layout: lane<16 -> col, rows v; lane>=16 -> same col, rows 8+v.
    #pragma unroll
    for (int ns = 0; ns < 2; ++ns) {
      v8f ct = ns ? accT1 : accT0;
      v8f ch = ns ? accH1 : accH0;
      const int col = ns ? c1 : c0;
      const float bTv = biasT[col];
      const float bHv = biasH[col];
      float ssum = 0.0f;
      #pragma unroll
      for (int v = 0; v < 8; ++v) {
        const int rrel = mrel + lhalf * 8 + v;
        const unsigned pk = sU[rrel * 256 + (col >> 1)];
        const float x = bf2f((unsigned short)((col & 1) ? (pk >> 16) : (pk & 0xFFFFu)));
        const float tau = 1.0f / (1.0f + __expf(-(ct[v] + bTv)));
        const float h   = fmaxf(ch[v] + bHv, 0.0f);
        ssum += h * tau + x * (1.0f - tau);
      }
      ssum += __shfl_down(ssum, 16);
      if (lane < 16) atomicAdd(&I[(size_t)b * E_DIM + col], ssum);
    }
  }
}

// ---------------------------------------------------------------------------
// Kernel 2: critic head  f[r] = lrelu(I[r] @ Wc1^T + bc1) @ Wc2^T + bc2
// ---------------------------------------------------------------------------
__global__ __launch_bounds__(256)
void critic_head(const float* __restrict__ I,
                 const float* __restrict__ Wc1, const float* __restrict__ bc1,
                 const float* __restrict__ Wc2, const float* __restrict__ bc2,
                 float* __restrict__ f) {
  __shared__ float sI[E_DIM];
  __shared__ float sT[DIM];
  const int r = blockIdx.x;
  const int j = threadIdx.x;
  sI[j]       = I[(size_t)r * E_DIM + j];
  sI[j + 256] = I[(size_t)r * E_DIM + j + 256];
  __syncthreads();

  float acc = bc1[j];
  const float* w = Wc1 + (size_t)j * E_DIM;
  #pragma unroll 4
  for (int k = 0; k < E_DIM; ++k) acc = fmaf(sI[k], w[k], acc);
  sT[j] = (acc > 0.0f) ? acc : NEG_S * acc;
  __syncthreads();

  float acc2 = bc2[j];
  const float* w2 = Wc2 + (size_t)j * DIM;
  #pragma unroll 4
  for (int k = 0; k < DIM; ++k) acc2 = fmaf(sT[k], w2[k], acc2);
  f[(size_t)r * DIM + j] = acc2;
}

// ---------------------------------------------------------------------------
// Kernel 3: mse row + cost.  (diff is independent of j: f_Q rows identical)
// ---------------------------------------------------------------------------
__global__ __launch_bounds__(256)
void pair_stats(const float* __restrict__ f,
                float* __restrict__ mse, float* __restrict__ cost) {
  __shared__ float red[256];
  const int i = blockIdx.x, e = threadIdx.x;
  const float d = f[(size_t)i * DIM + e] - f[(size_t)NB * DIM + e];
  const float m = d * d;
  mse[(size_t)i * DIM + e] = m;
  red[e] = m; __syncthreads();
  for (int s = 128; s > 0; s >>= 1) { if (e < s) red[e] += red[e + s]; __syncthreads(); }
  if (e == 0) cost[i] = red[0];
}

// ---------------------------------------------------------------------------
// Kernel 4: navigator MLP on the 512 distinct rows (N^2 collapses to N).
// ---------------------------------------------------------------------------
__global__ __launch_bounds__(256)
void navigator(const float* __restrict__ mse,
               const float* __restrict__ Wn1, const float* __restrict__ bn1,
               const float* __restrict__ Wn2, const float* __restrict__ bn2,
               const float* __restrict__ Wn3, const float* __restrict__ bn3,
               float* __restrict__ dist) {
  __shared__ float sm[DIM];
  __shared__ float sh[HID];
  __shared__ float red[256];
  const int i = blockIdx.x, t = threadIdx.x;
  sm[t] = mse[(size_t)i * DIM + t];
  __syncthreads();

  #pragma unroll
  for (int oo = 0; oo < 2; ++oo) {
    const int o = t + oo * 256;
    float acc = bn1[o];
    const float* w = Wn1 + (size_t)o * DIM;
    #pragma unroll 4
    for (int k = 0; k < DIM; ++k) acc = fmaf(sm[k], w[k], acc);
    sh[o] = (acc > 0.0f) ? acc : NEG_S * acc;
  }
  __syncthreads();

  float acc = bn2[t];
  const float* w = Wn2 + (size_t)t * HID;
  #pragma unroll 4
  for (int k = 0; k < HID; ++k) acc = fmaf(sh[k], w[k], acc);
  acc = (acc > 0.0f) ? acc : NEG_S * acc;
  red[t] = acc * Wn3[t];
  __syncthreads();
  for (int s = 128; s > 0; s >>= 1) { if (t < s) red[t] += red[t + s]; __syncthreads(); }
  if (t == 0) dist[i] = -(red[0] + bn3[0]);
}

// ---------------------------------------------------------------------------
// Kernel 5: loss = RHO*mean(cost) + (1-RHO)*sum(cost*softmax(dist))
// ---------------------------------------------------------------------------
__global__ __launch_bounds__(256)
void loss_kernel(const float* __restrict__ cost, const float* __restrict__ dist,
                 float* __restrict__ out) {
  __shared__ float r0[256], r1[256], r2[256];
  const int t = threadIdx.x;
  const float c0 = cost[t], c1 = cost[t + 256];
  const float d0 = dist[t], d1 = dist[t + 256];

  r0[t] = fmaxf(d0, d1); __syncthreads();
  for (int s = 128; s > 0; s >>= 1) { if (t < s) r0[t] = fmaxf(r0[t], r0[t + s]); __syncthreads(); }
  const float m = r0[0]; __syncthreads();

  const float e0 = __expf(d0 - m), e1 = __expf(d1 - m);
  r0[t] = e0 + e1;
  r1[t] = c0 * e0 + c1 * e1;
  r2[t] = c0 + c1;
  __syncthreads();
  for (int s = 128; s > 0; s >>= 1) {
    if (t < s) { r0[t] += r0[t + s]; r1[t] += r1[t + s]; r2[t] += r2[t + s]; }
    __syncthreads();
  }
  if (t == 0) out[0] = RHO_C * (r2[0] / (float)NB) + (1.0f - RHO_C) * (r1[0] / r0[0]);
}

// ---------------------------------------------------------------------------
extern "C" void kernel_launch(void* const* d_in, const int* in_sizes, int n_in,
                              void* d_out, int out_size, void* d_ws, size_t ws_size,
                              hipStream_t stream) {
  const float* K   = (const float*)d_in[0];
  const float* Q   = (const float*)d_in[1];
  const float* WT  = (const float*)d_in[2];
  const float* bT  = (const float*)d_in[3];
  const float* WH  = (const float*)d_in[4];
  const float* bH  = (const float*)d_in[5];
  const float* Wc1 = (const float*)d_in[6];
  const float* bc1 = (const float*)d_in[7];
  const float* Wc2 = (const float*)d_in[8];
  const float* bc2 = (const float*)d_in[9];
  const float* Wn1 = (const float*)d_in[10];
  const float* bn1 = (const float*)d_in[11];
  const float* Wn2 = (const float*)d_in[12];
  const float* bn2 = (const float*)d_in[13];
  const float* Wn3 = (const float*)d_in[14];
  const float* bn3 = (const float*)d_in[15];

  char* ws = (char*)d_ws;
  unsigned short* wtb = (unsigned short*)(ws);                  // 512 KB bf16 WT
  unsigned short* whb = (unsigned short*)(ws + 512 * 1024);     // 512 KB bf16 WH
  float* I    = (float*)(ws + 1024 * 1024);                     // 513*512 f32
  float* f    = (float*)((char*)I    + (size_t)513 * 512 * 4);  // 513*256 f32
  float* mse  = (float*)((char*)f    + (size_t)513 * 256 * 4);  // 512*256 f32
  float* cost = (float*)((char*)mse  + (size_t)512 * 256 * 4);  // 512 f32
  float* dist = (float*)((char*)cost + (size_t)512 * 4);        // 512 f32

  cvt_weights<<<512, 256, 0, stream>>>(WT, WH, wtb, whb, I, 513 * 512);
  critic_phase1<<<dim3(513, 8), 256, 0, stream>>>(K, Q, wtb, whb, bT, bH, I);
  critic_head<<<513, 256, 0, stream>>>(I, Wc1, bc1, Wc2, bc2, f);
  pair_stats<<<512, 256, 0, stream>>>(f, mse, cost);
  navigator<<<512, 256, 0, stream>>>(mse, Wn1, bn1, Wn2, bn2, Wn3, bn3, dist);
  loss_kernel<<<1, 256, 0, stream>>>(cost, dist, (float*)d_out);
}
